// Cae_62053687492914
// MI455X (gfx1250) — matrix operand order repaired and checked by reading, not compiled, not run on hardware
//
#include <hip/hip_runtime.h>
#include <math.h>

// ---------------- problem constants ----------------
constexpr int kB  = 128;          // batch
constexpr int kS  = 128;          // seq len
constexpr int kD  = 300;          // model dim
constexpr int kDp = 320;          // D padded to 32 (WMMA K granularity)
constexpr int kC  = 10;           // categories
constexpr int kP  = 3;            // polarity classes
constexpr int kH  = 150;          // per-direction hidden
constexpr int kHp = 160;          // H padded to 32
constexpr int kG  = 600;          // 4H
constexpr int kGp = 608;          // 4H padded to 16
constexpr int kBS = kB * kS;      // 16384 rows

typedef __attribute__((ext_vector_type(16))) __bf16 v16bf;
typedef __attribute__((ext_vector_type(8)))  float  v8f;

// ---------------- WMMA fragment loaders (CDNA5 ISA §7.12.2) ----------------
// A: 16x32 (MxK) bf16, row-major src, element (m,k) at base[m*lda + k].
__device__ inline v16bf load_a_frag(const __bf16* base, int lda, int lane) {
    int m = lane & 15, half = lane >> 4;
    v16bf a;
#pragma unroll
    for (int v = 0; v < 8; ++v) {
        int kb = (v < 4) ? (8 * half + 2 * v) : (16 + 8 * half + 2 * (v - 4));
        a[2 * v]     = base[m * lda + kb];
        a[2 * v + 1] = base[m * lda + kb + 1];
    }
    return a;
}
// B: 32x16 (KxN) bf16 stored transposed [N][K]: element (k,n) at base[n*ldb + k].
__device__ inline v16bf load_b_frag(const __bf16* base, int ldb, int lane) {
    int n = lane & 15, half = lane >> 4;
    v16bf b;
#pragma unroll
    for (int v = 0; v < 8; ++v) {
        int kb = (v < 4) ? (8 * half + 2 * v) : (16 + 8 * half + 2 * (v - 4));
        b[2 * v]     = base[n * ldb + kb];
        b[2 * v + 1] = base[n * ldb + kb + 1];
    }
    return b;
}
// C/D 16x16 f32: VGPR r -> row m = r + 8*half, col n = lane&15.

__device__ inline float sigmoidf_(float x) { return 1.f / (1.f + expf(-x)); }

// Async-copy a contiguous block (16B granules) global -> LDS via CDNA5
// GLOBAL_LOAD_ASYNC_TO_LDS_B128 (ASYNCcnt, CDNA5 ISA).
// LDS address of a __shared__ object is the low 32 bits of its generic address.
__device__ inline void stage_async(const __bf16* __restrict__ src, __bf16* dstLDS,
                                   int bytes, int tid, int nthreads) {
    unsigned lds0 = (unsigned)(size_t)dstLDS;
    unsigned long long g0 = (unsigned long long)(size_t)src;
    for (int i = tid; i < bytes / 16; i += nthreads) {
        unsigned l = lds0 + (unsigned)i * 16u;
        unsigned long long g = g0 + (unsigned long long)i * 16ull;
        asm volatile("global_load_async_to_lds_b128 %0, %1, off"
                     :: "v"(l), "v"(g) : "memory");
    }
    asm volatile("s_wait_asynccnt 0x0" ::: "memory");
}

// ---------------- prep kernels ----------------
__global__ void embed_kernel(const int* __restrict__ tokens, const float* __restrict__ emb,
                             float* __restrict__ we, __bf16* __restrict__ webf) {
    size_t idx = (size_t)blockIdx.x * 256 + threadIdx.x;
    if (idx >= (size_t)kBS * kDp) return;
    int bs = (int)(idx / kDp), dp = (int)(idx % kDp);
    float v = 0.f;
    if (dp < kD) {
        int tok = tokens[bs];
        v = emb[(size_t)tok * kD + dp];
        we[(size_t)bs * kD + dp] = v;
    }
    webf[idx] = (__bf16)v;
}

// W: [batch][K][N] f32 -> WT: [batch][Np][Kp] bf16 (zero padded, transposed)
__global__ void packT_kernel(const float* __restrict__ W, __bf16* __restrict__ WT,
                             int batch, int K, int N, int Kp, int Np) {
    size_t total = (size_t)batch * Np * Kp;
    size_t idx = (size_t)blockIdx.x * 256 + threadIdx.x;
    if (idx >= total) return;
    int c = (int)(idx / ((size_t)Np * Kp));
    int r = (int)(idx % ((size_t)Np * Kp));
    int n = r / Kp, k = r % Kp;
    float v = (k < K && n < N) ? W[((size_t)c * K + k) * N + n] : 0.f;
    WT[((size_t)c * Np + n) * Kp + k] = (__bf16)v;
}

// vec: [batch][N] f32 -> [batch][Np] f32 zero-padded (branch-free GEMM epilogues)
__global__ void pack_vec_kernel(const float* __restrict__ src, float* __restrict__ dst,
                                int batch, int N, int Np) {
    size_t idx = (size_t)blockIdx.x * 256 + threadIdx.x;
    if (idx >= (size_t)batch * Np) return;
    int c = (int)(idx / Np), n = (int)(idx % Np);
    dst[idx] = (n < N) ? src[(size_t)c * N + n] : 0.f;
}

__global__ void zero_bf16_kernel(__bf16* p, size_t n) {
    size_t idx = (size_t)blockIdx.x * 256 + threadIdx.x;
    if (idx < n) p[idx] = (__bf16)0.f;
}

// ---------------- fused attention sims (async-LDS A panel + 2xN-blocked WMMA) ----------------
// sims[c,b,s] = sum_e tanh( (h @ We[c])[bs,e] + be[c,e] ) * ue[c,e]   (masked -> -inf)
__global__ void __launch_bounds__(128)
attn_sims_kernel(const __bf16* __restrict__ Xp,   // [kBS][kDp]
                 const __bf16* __restrict__ WT,   // [kC][kDp][kDp] (N-major)
                 const float* __restrict__ bep,   // [kC][kDp] padded
                 const float* __restrict__ uep,   // [kC][kDp] padded
                 const unsigned char* __restrict__ mask,
                 float* __restrict__ sims) {      // [kC][kBS]
    __shared__ __attribute__((aligned(16))) __bf16 As[64 * kDp];  // 40 KB A panel
    int tid = threadIdx.x;
    int wave = tid >> 5, lane = tid & 31;
    int half = lane >> 4, n = lane & 15;
    int rowblk = blockIdx.x * 64;

    stage_async(Xp + (size_t)rowblk * kDp, As, 64 * kDp * 2, tid, 128);
    __syncthreads();

    const __bf16* Aw = As + wave * 16 * kDp;
    for (int c = 0; c < kC; ++c) {
        const __bf16* WTc = WT + (size_t)c * kDp * kDp;
        float partial[8] = {0.f, 0.f, 0.f, 0.f, 0.f, 0.f, 0.f, 0.f};
        for (int ntp = 0; ntp < kDp / 32; ++ntp) {
            v8f acc0 = {}, acc1 = {};
#pragma unroll 2
            for (int kc = 0; kc < kDp / 32; ++kc) {
                v16bf a  = load_a_frag(Aw + kc * 32, kDp, lane);
                v16bf b0 = load_b_frag(WTc + (size_t)(ntp * 32) * kDp + kc * 32, kDp, lane);
                v16bf b1 = load_b_frag(WTc + (size_t)(ntp * 32 + 16) * kDp + kc * 32, kDp, lane);
                acc0 = __builtin_amdgcn_wmma_f32_16x16x32_bf16(false, a, false, b0, (short)0,
                                                               acc0, false, false);
                acc1 = __builtin_amdgcn_wmma_f32_16x16x32_bf16(false, a, false, b1, (short)0,
                                                               acc1, false, false);
            }
            int e0 = ntp * 32 + n, e1 = e0 + 16;
            float be0 = bep[c * kDp + e0], ue0 = uep[c * kDp + e0];
            float be1 = bep[c * kDp + e1], ue1 = uep[c * kDp + e1];
#pragma unroll
            for (int r = 0; r < 8; ++r)
                partial[r] += tanhf(acc0[r] + be0) * ue0 + tanhf(acc1[r] + be1) * ue1;
        }
#pragma unroll
        for (int r = 0; r < 8; ++r) {
            float v = partial[r];
            v += __shfl_xor(v, 1, 32);
            v += __shfl_xor(v, 2, 32);
            v += __shfl_xor(v, 4, 32);
            v += __shfl_xor(v, 8, 32);
            if (n == 0) {
                int row = rowblk + wave * 16 + r + 8 * half;   // row = b*kS + s
                sims[(size_t)c * kBS + row] = mask[row] ? v : -__builtin_huge_valf();
            }
        }
        __syncthreads();   // keep As coherent view trivial; also paces waves
    }
}

// ---------------- softmax over S (in-place, rows = C*B) ----------------
__global__ void softmax128_kernel(float* __restrict__ x) {
    __shared__ float red[128];
    int row = blockIdx.x, tid = threadIdx.x;
    float v = x[(size_t)row * kS + tid];
    red[tid] = v;
    __syncthreads();
    for (int off = 64; off > 0; off >>= 1) {
        if (tid < off) red[tid] = fmaxf(red[tid], red[tid + off]);
        __syncthreads();
    }
    float m = red[0];
    __syncthreads();
    float e = expf(v - m);   // -inf -> 0
    red[tid] = e;
    __syncthreads();
    for (int off = 64; off > 0; off >>= 1) {
        if (tid < off) red[tid] += red[tid + off];
        __syncthreads();
    }
    x[(size_t)row * kS + tid] = e / red[0];
}

// out[c,b,:] = sum_s w[c,b,s] * h[b,s,:]
__global__ void wsum_kernel(const float* __restrict__ w, const float* __restrict__ h,
                            float* __restrict__ out) {
    int cb = blockIdx.x, b = cb % kB;
    for (int d = threadIdx.x; d < kD; d += blockDim.x) {
        float acc = 0.f;
        const float* wr = w + (size_t)cb * kS;
        for (int s = 0; s < kS; ++s) acc += wr[s] * h[((size_t)b * kS + s) * kD + d];
        out[(size_t)cb * kD + d] = acc;
    }
}

// t[c,b,s] = cat[c,b,:] . h[b,s,:]
__global__ void dot_kernel(const float* __restrict__ cat, const float* __restrict__ h,
                           float* __restrict__ t) {
    int cb = blockIdx.x, b = cb % kB, s = threadIdx.x;
    const float* cr = cat + (size_t)cb * kD;
    const float* hr = h + ((size_t)b * kS + s) * kD;
    float acc = 0.f;
    for (int d = 0; d < kD; ++d) acc += cr[d] * hr[d];
    t[(size_t)cb * kS + s] = acc;
}

// ---------------- LSTM input projection: xs = x @ Wi + b (async-LDS A + WMMA) ----------------
__global__ void __launch_bounds__(128)
lstm_xs_kernel(const __bf16* __restrict__ Xp, const __bf16* __restrict__ WiTf,
               const __bf16* __restrict__ WiTb, const float* __restrict__ bfp,
               const float* __restrict__ bbp, float* __restrict__ xs) {
    __shared__ __attribute__((aligned(16))) __bf16 As[64 * kDp];  // 40 KB A panel
    int tid = threadIdx.x;
    int wave = tid >> 5, lane = tid & 31;
    int half = lane >> 4, n = lane & 15;
    int rowblk = blockIdx.x * 64;

    stage_async(Xp + (size_t)rowblk * kDp, As, 64 * kDp * 2, tid, 128);
    __syncthreads();

    const __bf16* Aw = As + wave * 16 * kDp;
    for (int dir = 0; dir < 2; ++dir) {
        const __bf16* WiT  = dir ? WiTb : WiTf;
        const float*  bias = dir ? bbp : bfp;   // [kGp] padded
        for (int ntp = 0; ntp < kGp / 32; ++ntp) {
            v8f acc0 = {}, acc1 = {};
#pragma unroll 2
            for (int kc = 0; kc < kDp / 32; ++kc) {
                v16bf a  = load_a_frag(Aw + kc * 32, kDp, lane);
                v16bf b0 = load_b_frag(WiT + (size_t)(ntp * 32) * kDp + kc * 32, kDp, lane);
                v16bf b1 = load_b_frag(WiT + (size_t)(ntp * 32 + 16) * kDp + kc * 32, kDp, lane);
                acc0 = __builtin_amdgcn_wmma_f32_16x16x32_bf16(false, a, false, b0, (short)0,
                                                               acc0, false, false);
                acc1 = __builtin_amdgcn_wmma_f32_16x16x32_bf16(false, a, false, b1, (short)0,
                                                               acc1, false, false);
            }
            int e0 = ntp * 32 + n, e1 = e0 + 16;
            float bi0 = bias[e0], bi1 = bias[e1];
#pragma unroll
            for (int r = 0; r < 8; ++r) {
                int row = rowblk + wave * 16 + r + 8 * half;
                size_t base = ((size_t)dir * kBS + row) * kGp;
                xs[base + e0] = acc0[r] + bi0;
                xs[base + e1] = acc1[r] + bi1;
            }
        }
    }
}

// ---------------- LSTM recurrence: per-step 16x608x160 WMMA GEMM in LDS ----------------
__global__ void __launch_bounds__(256) lstm_rec_kernel(const float* __restrict__ xs,
                                                       const __bf16* __restrict__ WhTf,
                                                       const __bf16* __restrict__ WhTb,
                                                       float* __restrict__ lstm,
                                                       __bf16* __restrict__ lstmbf) {
    __shared__ __bf16 hS[16 * kHp];   // h state, bf16, K-padded (zeros in pad)
    __shared__ float  cS[16 * kH];    // c state
    __shared__ float  gS[16 * kGp];   // gate pre-activations for this step
    int dir = blockIdx.y;
    int b0 = blockIdx.x * 16;
    const __bf16* WhT = dir ? WhTb : WhTf;
    int tid = threadIdx.x, wave = tid >> 5, lane = tid & 31;
    int half = lane >> 4, n = lane & 15;
    for (int i = tid; i < 16 * kHp; i += 256) hS[i] = (__bf16)0.f;
    for (int i = tid; i < 16 * kH; i += 256) cS[i] = 0.f;
    __syncthreads();
    for (int step = 0; step < kS; ++step) {
        int t = dir ? (kS - 1 - step) : step;
        for (int ntp = wave; ntp < kGp / 32; ntp += 8) {
            v8f acc0 = {}, acc1 = {};
#pragma unroll
            for (int kc = 0; kc < kHp / 32; ++kc) {
                v16bf a  = load_a_frag(hS + kc * 32, kHp, lane);
                v16bf b0 = load_b_frag(WhT + (size_t)(ntp * 32) * kHp + kc * 32, kHp, lane);
                v16bf b1 = load_b_frag(WhT + (size_t)(ntp * 32 + 16) * kHp + kc * 32, kHp, lane);
                acc0 = __builtin_amdgcn_wmma_f32_16x16x32_bf16(false, a, false, b0, (short)0,
                                                               acc0, false, false);
                acc1 = __builtin_amdgcn_wmma_f32_16x16x32_bf16(false, a, false, b1, (short)0,
                                                               acc1, false, false);
            }
#pragma unroll
            for (int r = 0; r < 8; ++r) {
                int m = r + 8 * half;
                int e0 = ntp * 32 + n, e1 = e0 + 16;
                size_t xb = ((size_t)dir * kBS + (size_t)(b0 + m) * kS + t) * kGp;
                gS[m * kGp + e0] = acc0[r] + xs[xb + e0];
                gS[m * kGp + e1] = acc1[r] + xs[xb + e1];
            }
        }
        __syncthreads();
        for (int u = tid; u < 16 * kH; u += 256) {
            int m = u / kH, j = u % kH;
            float gi = gS[m * kGp + j];
            float gf = gS[m * kGp + kH + j];
            float gg = gS[m * kGp + 2 * kH + j];
            float go = gS[m * kGp + 3 * kH + j];
            float cc = sigmoidf_(gf) * cS[u] + sigmoidf_(gi) * tanhf(gg);
            float hh = sigmoidf_(go) * tanhf(cc);
            cS[u] = cc;
            size_t pos = (size_t)(b0 + m) * kS + t;
            lstm[pos * kD + dir * kH + j]    = hh;
            lstmbf[pos * kDp + dir * kH + j] = (__bf16)hh;
            hS[m * kHp + j] = (__bf16)hh;
        }
        __syncthreads();
    }
}

// ---------------- MLP heads: relu(x@w1+b1)@w2+b2, x = concat(xa, xb) ----------------
__global__ void head_kernel(const float* __restrict__ xa, const float* __restrict__ xb,
                            const float* __restrict__ w1, const float* __restrict__ b1,
                            const float* __restrict__ w2, const float* __restrict__ b2,
                            float* __restrict__ out, int Kout) {
    int cb = blockIdx.x, c = cb / kB;
    int k = threadIdx.x;  // 32 hidden units, one wave
    const float* w1c = w1 + (size_t)c * kG * 32;
    const float* xav = xa + (size_t)cb * kD;
    const float* xbv = xb + (size_t)cb * kD;
    float h = b1[c * 32 + k];
    for (int d = 0; d < kD; ++d) h += xav[d] * w1c[d * 32 + k];
    for (int d = 0; d < kD; ++d) h += xbv[d] * w1c[(kD + d) * 32 + k];
    h = fmaxf(h, 0.f);
    for (int o = 0; o < Kout; ++o) {
        float p = h * w2[((size_t)c * 32 + k) * Kout + o];
        p += __shfl_xor(p, 16, 32);
        p += __shfl_xor(p, 8, 32);
        p += __shfl_xor(p, 4, 32);
        p += __shfl_xor(p, 2, 32);
        p += __shfl_xor(p, 1, 32);
        if (k == 0) out[(size_t)cb * Kout + o] = p + b2[c * Kout + o];
    }
}

// ---------------- predictions + losses ----------------
__global__ void final_kernel(const float* __restrict__ det, const float* __restrict__ sen,
                             const int* __restrict__ labels, float* __restrict__ out) {
    __shared__ float cb_s[kC], cb_c[kC], sn_s[kC], sn_c[kC];
    int tid = threadIdx.x;
    if (tid < kC) { cb_s[tid] = 0.f; cb_c[tid] = 0.f; sn_s[tid] = 0.f; sn_c[tid] = 0.f; }
    __syncthreads();
    for (int idx = tid; idx < kC * kB; idx += 256) {
        int c = idx / kB, b = idx % kB;
        float x = det[idx];
        out[idx] = 1.f / (1.f + expf(-x));
        int cl = labels[b * 2 * kC + c];
        if (cl != -1) {
            float y = (float)(cl > 0 ? cl : 0);
            float bce = fmaxf(x, 0.f) - x * y + log1pf(expf(-fabsf(x)));
            atomicAdd(&cb_s[c], bce);
            atomicAdd(&cb_c[c], 1.f);
        }
        float l0 = sen[idx * kP + 0], l1 = sen[idx * kP + 1], l2 = sen[idx * kP + 2];
        float mx = fmaxf(l0, fmaxf(l1, l2));
        float e0 = expf(l0 - mx), e1 = expf(l1 - mx), e2 = expf(l2 - mx);
        float se = e0 + e1 + e2;
        out[kC * kB + idx * kP + 0] = e0 / se;
        out[kC * kB + idx * kP + 1] = e1 / se;
        out[kC * kB + idx * kP + 2] = e2 / se;
        int pl = labels[b * 2 * kC + kC + c];
        if (pl != -1) {
            int lab = pl > 0 ? pl : 0;
            float lse = mx + logf(se);
            float lv = (lab == 0) ? l0 : ((lab == 1) ? l1 : l2);
            atomicAdd(&sn_s[c], lse - lv);
            atomicAdd(&sn_c[c], 1.f);
        }
    }
    __syncthreads();
    if (tid == 0) {
        float loss = 0.f;
        for (int c = 0; c < kC; ++c) {
            if (cb_c[c] > 0.f) loss += cb_s[c] / cb_c[c];
            if (sn_c[c] > 0.f) loss += sn_s[c] / sn_c[c];
        }
        out[kC * kB + kC * kB * kP] = loss;
    }
}

// ---------------- host orchestration ----------------
static inline int cdiv(size_t a, int b) { return (int)((a + b - 1) / b); }

extern "C" void kernel_launch(void* const* d_in, const int* in_sizes, int n_in,
                              void* d_out, int out_size, void* d_ws, size_t ws_size,
                              hipStream_t stream) {
    (void)in_sizes; (void)n_in; (void)out_size; (void)ws_size;
    const int*           tokens = (const int*)d_in[0];
    const int*           labels = (const int*)d_in[1];
    const unsigned char* mask   = (const unsigned char*)d_in[2];   // jnp bool = 1 byte
    const float* emb  = (const float*)d_in[3];
    const float* We   = (const float*)d_in[4];
    const float* be   = (const float*)d_in[5];
    const float* ue   = (const float*)d_in[6];
    const float* Wl   = (const float*)d_in[7];
    const float* blb  = (const float*)d_in[8];
    const float* ul   = (const float*)d_in[9];
    const float* Wi_f = (const float*)d_in[10];
    const float* Wh_f = (const float*)d_in[11];
    const float* b_f  = (const float*)d_in[12];
    const float* Wi_b = (const float*)d_in[13];
    const float* Wh_b = (const float*)d_in[14];
    const float* b_b  = (const float*)d_in[15];
    const float* dw1 = (const float*)d_in[16];
    const float* db1 = (const float*)d_in[17];
    const float* dw2 = (const float*)d_in[18];
    const float* db2 = (const float*)d_in[19];
    const float* sw1 = (const float*)d_in[20];
    const float* sb1 = (const float*)d_in[21];
    const float* sw2 = (const float*)d_in[22];
    const float* sb2 = (const float*)d_in[23];
    float* out = (float*)d_out;

    // workspace carve (deterministic, ~150 MB)
    size_t off = 0;
    auto alloc = [&](size_t bytes) -> void* {
        void* p = (char*)d_ws + off;
        off = (off + bytes + 255) & ~(size_t)255;
        return p;
    };
    float*  we      = (float*)alloc((size_t)kBS * kD * 4);
    __bf16* webf    = (__bf16*)alloc((size_t)kBS * kDp * 2);
    __bf16* WeT     = (__bf16*)alloc((size_t)kC * kDp * kDp * 2);
    __bf16* WlT     = (__bf16*)alloc((size_t)kC * kDp * kDp * 2);
    __bf16* WiTf    = (__bf16*)alloc((size_t)kGp * kDp * 2);
    __bf16* WiTb    = (__bf16*)alloc((size_t)kGp * kDp * 2);
    __bf16* WhTf    = (__bf16*)alloc((size_t)kGp * kHp * 2);
    __bf16* WhTb    = (__bf16*)alloc((size_t)kGp * kHp * 2);
    float*  bep     = (float*)alloc((size_t)kC * kDp * 4);
    float*  uep     = (float*)alloc((size_t)kC * kDp * 4);
    float*  blp     = (float*)alloc((size_t)kC * kDp * 4);
    float*  ulp     = (float*)alloc((size_t)kC * kDp * 4);
    float*  bfp     = (float*)alloc((size_t)kGp * 4);
    float*  bbp     = (float*)alloc((size_t)kGp * 4);
    float*  xs      = (float*)alloc((size_t)2 * kBS * kGp * 4);   // ~80 MB
    float*  lstm    = (float*)alloc((size_t)kBS * kD * 4);
    __bf16* lstmbf  = (__bf16*)alloc((size_t)kBS * kDp * 2);
    float*  sims    = (float*)alloc((size_t)kC * kBS * 4);        // reused as alpha/sa
    float*  cat_e   = (float*)alloc((size_t)kC * kB * kD * 4);
    float*  sent_e  = (float*)alloc((size_t)kC * kB * kD * 4);
    float*  cat_l   = (float*)alloc((size_t)kC * kB * kD * 4);
    float*  sent_l  = (float*)alloc((size_t)kC * kB * kD * 4);
    float*  det_lg  = (float*)alloc((size_t)kC * kB * 4);
    float*  sen_lg  = (float*)alloc((size_t)kC * kB * kP * 4);

    // ---- prep ----
    embed_kernel<<<cdiv((size_t)kBS * kDp, 256), 256, 0, stream>>>(tokens, emb, we, webf);
    packT_kernel<<<cdiv((size_t)kC * kDp * kDp, 256), 256, 0, stream>>>(We, WeT, kC, kD, kD, kDp, kDp);
    packT_kernel<<<cdiv((size_t)kC * kDp * kDp, 256), 256, 0, stream>>>(Wl, WlT, kC, kD, kD, kDp, kDp);
    packT_kernel<<<cdiv((size_t)kGp * kDp, 256), 256, 0, stream>>>(Wi_f, WiTf, 1, kD, kG, kDp, kGp);
    packT_kernel<<<cdiv((size_t)kGp * kDp, 256), 256, 0, stream>>>(Wi_b, WiTb, 1, kD, kG, kDp, kGp);
    packT_kernel<<<cdiv((size_t)kGp * kHp, 256), 256, 0, stream>>>(Wh_f, WhTf, 1, kH, kG, kHp, kGp);
    packT_kernel<<<cdiv((size_t)kGp * kHp, 256), 256, 0, stream>>>(Wh_b, WhTb, 1, kH, kG, kHp, kGp);
    pack_vec_kernel<<<cdiv((size_t)kC * kDp, 256), 256, 0, stream>>>(be, bep, kC, kD, kDp);
    pack_vec_kernel<<<cdiv((size_t)kC * kDp, 256), 256, 0, stream>>>(ue, uep, kC, kD, kDp);
    pack_vec_kernel<<<cdiv((size_t)kC * kDp, 256), 256, 0, stream>>>(blb, blp, kC, kD, kDp);
    pack_vec_kernel<<<cdiv((size_t)kC * kDp, 256), 256, 0, stream>>>(ul, ulp, kC, kD, kDp);
    pack_vec_kernel<<<cdiv((size_t)kGp, 256), 256, 0, stream>>>(b_f, bfp, 1, kG, kGp);
    pack_vec_kernel<<<cdiv((size_t)kGp, 256), 256, 0, stream>>>(b_b, bbp, 1, kG, kGp);
    zero_bf16_kernel<<<cdiv((size_t)kBS * kDp, 256), 256, 0, stream>>>(lstmbf, (size_t)kBS * kDp);

    int rowsCB = kC * kB;

    // ---- stage 1 (embeddings) ----
    attn_sims_kernel<<<kBS / 64, 128, 0, stream>>>(webf, WeT, bep, uep, mask, sims);
    softmax128_kernel<<<rowsCB, 128, 0, stream>>>(sims);                 // alpha
    wsum_kernel<<<rowsCB, 128, 0, stream>>>(sims, we, cat_e);            // cat_e
    dot_kernel<<<rowsCB, 128, 0, stream>>>(cat_e, we, sims);             // cat·h
    softmax128_kernel<<<rowsCB, 128, 0, stream>>>(sims);                 // sa (unmasked)
    wsum_kernel<<<rowsCB, 128, 0, stream>>>(sims, we, sent_e);           // sent_e

    // ---- biLSTM ----
    lstm_xs_kernel<<<kBS / 64, 128, 0, stream>>>(webf, WiTf, WiTb, bfp, bbp, xs);
    dim3 gRec(kB / 16, 2, 1);
    lstm_rec_kernel<<<gRec, 256, 0, stream>>>(xs, WhTf, WhTb, lstm, lstmbf);

    // ---- stage 2 (LSTM output) ----
    attn_sims_kernel<<<kBS / 64, 128, 0, stream>>>(lstmbf, WlT, blp, ulp, mask, sims);
    softmax128_kernel<<<rowsCB, 128, 0, stream>>>(sims);
    wsum_kernel<<<rowsCB, 128, 0, stream>>>(sims, lstm, cat_l);
    dot_kernel<<<rowsCB, 128, 0, stream>>>(cat_l, lstm, sims);
    softmax128_kernel<<<rowsCB, 128, 0, stream>>>(sims);
    wsum_kernel<<<rowsCB, 128, 0, stream>>>(sims, lstm, sent_l);

    // ---- heads + losses ----
    head_kernel<<<rowsCB, 32, 0, stream>>>(cat_e, cat_l, dw1, db1, dw2, db2, det_lg, 1);
    head_kernel<<<rowsCB, 32, 0, stream>>>(sent_e, sent_l, sw1, sb1, sw2, sb2, sen_lg, kP);
    final_kernel<<<1, 256, 0, stream>>>(det_lg, sen_lg, labels, out);
}
